// Transformer_53712861004549
// MI455X (gfx1250) — matrix operand order repaired and checked by reading, not compiled, not run on hardware
//
#include <hip/hip_runtime.h>
#include <hip/hip_bf16.h>

// ---- model constants (fixed by the reference) ----
#define D_    512
#define H_    8
#define DH_   64
#define HID_  2048
#define L_    6
#define S_    512
#define T_    512
#define B_    4
#define MROWS (B_ * S_)   // 2048
#define VT_   16000

typedef __bf16 bf16_t;
typedef bf16_t v16bf __attribute__((ext_vector_type(16)));
typedef bf16_t v8bf  __attribute__((ext_vector_type(8)));
typedef float  v8f   __attribute__((ext_vector_type(8)));

typedef unsigned int u32x4 __attribute__((ext_vector_type(4)));
typedef int          i32x4 __attribute__((ext_vector_type(4)));
typedef int          i32x8 __attribute__((ext_vector_type(8)));

union Frag16 { v16bf v; v8bf h[2]; };

#if __has_builtin(__builtin_amdgcn_tensor_load_to_lds) && \
    __has_builtin(__builtin_amdgcn_s_wait_tensorcnt)
#define USE_TDM 1
#else
#define USE_TDM 0
#endif

__device__ __forceinline__ v8f wmma_bf16(const Frag16& a, const Frag16& b, v8f c) {
  return __builtin_amdgcn_wmma_f32_16x16x32_bf16(
      false, a.v, false, b.v, (short)0, c, false, false);
}

__device__ __forceinline__ unsigned lds_off(const void* p) {
  // low 32 bits of the flat address are the LDS byte offset (ISA §10.2 aperture)
  return (unsigned)(unsigned long long)(uintptr_t)p;
}

// TDM: load a 32(K) x 128(N) bf16 tile of Bt (row stride ldb elements) into LDS.
// LDS layout: 128 rows, 64B data + 16B pad => 80B (40-element) stride.
__device__ __forceinline__ void tdm_load_tile(const bf16_t* gsrc,
                                              unsigned ldsByteAddr,
                                              int ldbElems) {
#if USE_TDM
  unsigned long long ga = (unsigned long long)(uintptr_t)gsrc;
  u32x4 g0;
  g0[0] = 1u;                                         // count = 1 (valid D#)
  g0[1] = ldsByteAddr;                                // lds_addr
  g0[2] = (unsigned)(ga & 0xFFFFFFFFu);               // global_addr[31:0]
  g0[3] = (unsigned)((ga >> 32) & 0x01FFFFFFu)        // global_addr[56:32]
          | (2u << 30);                               // type = 2 (image)
  i32x8 g1;
  g1[0] = (int)((1u << 16)      // data_size = 1 (2 bytes)
              | (1u << 20)      // pad_enable
              | (3u << 22)      // pad_interval: every 16 DWORDs (64B)
              | (3u << 25));    // pad_amount: 4 DWORDs (16B)
  g1[1] = 0;                    // tensor_dim0 = 1<<20 -> low 16 bits = 0
  g1[2] = 0x10;                 // tensor_dim0 hi | tensor_dim1 lo (dim1 = 1<<20)
  g1[3] = (32 << 16) | 0x10;    // tensor_dim1 hi | tile_dim0 = 32
  g1[4] = 128;                  // tile_dim1 = 128, tile_dim2 = 0
  g1[5] = ldbElems;             // tensor_dim0_stride[31:0]
  g1[6] = 0;                    // stride hi | tensor_dim1_stride lo
  g1[7] = 0;
  i32x4 g2; g2[0] = 1 << 20; g2[1] = 1 << 20; g2[2] = 0; g2[3] = 0;
  i32x4 g3; g3[0] = 0; g3[1] = 0; g3[2] = 0; g3[3] = 0;
#if defined(__clang_major__) && (__clang_major__ >= 23)
  i32x8 gz = {};
  __builtin_amdgcn_tensor_load_to_lds(g0, g1, g2, g3, gz, 0);
#else
  __builtin_amdgcn_tensor_load_to_lds(g0, g1, g2, g3, 0);
#endif
#else
  (void)gsrc; (void)ldsByteAddr; (void)ldbElems;
#endif
}

// ---------------------------------------------------------------------------
// Shared GEMM argument block.
//   A : bf16 [M,K] row-major; Bt: bf16 [N,K] row-major (B transposed)
//   modes: 0 plain [M,N]; 1 -> [B,H,S,DH]; 2 -> [B,H,DH,S];
//          3 -> context merge [B*T, D], batched z=(b,h)
// ---------------------------------------------------------------------------
struct GemmArgs {
  const bf16_t* A;
  const bf16_t* Bt;
  const float*  bias;
  float*        Cf;
  bf16_t*       Cb;
  long sA, sB, sCf, sCb;   // per-z strides (elements)
  int  K, lda, ldb, ldc;
  int  mode, relu;
  float scale;
};

// ---------------------------------------------------------------------------
// Small-N GEMM (direct from global), block tile 128x64, wave tile 32x32.
// Used for the attention context GEMM (N = 64, mode 3).
// ---------------------------------------------------------------------------
__global__ __launch_bounds__(256)
void gemm_bf16_wmma(GemmArgs g) {
  const int tid  = threadIdx.x;
  const int wave = tid >> 5;
  const int lane = tid & 31;
  const int hl   = lane >> 4;
  const int l16  = lane & 15;
  const int wm   = wave >> 1;
  const int wn   = wave & 1;
  const long z   = blockIdx.z;

  const bf16_t* A  = g.A  + z * g.sA;
  const bf16_t* Bt = g.Bt + z * g.sB;

  const int m0 = blockIdx.x * 128 + wm * 32;
  const int n0 = blockIdx.y * 64  + wn * 32;

  v8f acc[2][2] = {};

  const bf16_t* aRow0 = A + (long)(m0 + l16)      * g.lda + hl * 16;
  const bf16_t* aRow1 = A + (long)(m0 + 16 + l16) * g.lda + hl * 16;
  const bf16_t* bCol0 = Bt + (long)(n0 + l16)      * g.ldb + hl * 16;
  const bf16_t* bCol1 = Bt + (long)(n0 + 16 + l16) * g.ldb + hl * 16;

  for (int kb = 0; kb < g.K; kb += 32) {
    __builtin_prefetch(aRow0 + kb + 128, 0, 1);
    __builtin_prefetch(bCol0 + kb + 128, 0, 1);

    Frag16 a0, a1, b0, b1;
    a0.h[0] = *(const v8bf*)(aRow0 + kb);
    a0.h[1] = *(const v8bf*)(aRow0 + kb + 8);
    a1.h[0] = *(const v8bf*)(aRow1 + kb);
    a1.h[1] = *(const v8bf*)(aRow1 + kb + 8);
    b0.v    = *(const v16bf*)(bCol0 + kb);
    b1.v    = *(const v16bf*)(bCol1 + kb);

    acc[0][0] = wmma_bf16(a0, b0, acc[0][0]);
    acc[0][1] = wmma_bf16(a0, b1, acc[0][1]);
    acc[1][0] = wmma_bf16(a1, b0, acc[1][0]);
    acc[1][1] = wmma_bf16(a1, b1, acc[1][1]);
  }

  #pragma unroll
  for (int i = 0; i < 2; i++) {
    #pragma unroll
    for (int j = 0; j < 2; j++) {
      const int nn = n0 + j * 16 + l16;
      const float bv = g.bias ? g.bias[nn] : 0.0f;
      #pragma unroll
      for (int r = 0; r < 8; r++) {
        const int mm = m0 + i * 16 + hl * 8 + r;
        float v = acc[i][j][r] * g.scale + bv;
        if (g.relu) v = v > 0.0f ? v : 0.0f;
        long idx;
        if (g.mode == 0) {
          idx = (long)mm * g.ldc + nn;
        } else if (g.mode == 1) {
          const int bb = mm >> 9, ss = mm & 511;
          const int hh = nn >> 6, dd = nn & 63;
          idx = (((long)(bb * H_ + hh) * S_) + ss) * DH_ + dd;
        } else if (g.mode == 2) {
          const int bb = mm >> 9, ss = mm & 511;
          const int hh = nn >> 6, dd = nn & 63;
          idx = (((long)(bb * H_ + hh) * DH_) + dd) * S_ + ss;
        } else {
          const int bb = (int)(blockIdx.z >> 3), hh = (int)(blockIdx.z & 7);
          idx = ((long)(bb * T_ + mm)) * D_ + (hh * DH_ + nn);
        }
        if (g.Cf) g.Cf[z * g.sCf + idx] = v;
        if (g.Cb) g.Cb[z * g.sCb + idx] = (bf16_t)v;
      }
    }
  }
}

// ---------------------------------------------------------------------------
// Main GEMM: block tile 128x128, wave tile 32x64 (8 WMMAs / K-step / wave).
// B tile (32K x 128N) staged in LDS via TDM tensor_load_to_lds, double
// buffered, s_wait_tensorcnt + barrier synchronized. A streamed from global.
// Requires M%128==0, N%128==0, K%32==0; modes 0/1/2.
// ---------------------------------------------------------------------------
#define LDS_STRIDE 40   // elements: 64B data + 16B pad per row

__global__ __launch_bounds__(256)
void gemm_bf16_wmma_lds(GemmArgs g) {
  __shared__ __align__(16) bf16_t smem[2][128 * LDS_STRIDE];

  const int tid  = threadIdx.x;
  const int wave = tid >> 5;
  const int lane = tid & 31;
  const int hl   = lane >> 4;
  const int l16  = lane & 15;
  const int wm   = wave >> 1;      // 0..3 : M
  const int wn   = wave & 1;       // 0..1 : N
  const long z   = blockIdx.z;

  const bf16_t* A  = g.A  + z * g.sA;
  const bf16_t* Bt = g.Bt + z * g.sB + (long)blockIdx.y * 128 * g.ldb;

  const int m0    = blockIdx.x * 128 + wm * 32;
  const int n0loc = wn * 64;

  v8f acc[2][4] = {};

  const bf16_t* aRow0 = A + (long)(m0 + l16)      * g.lda + hl * 16;
  const bf16_t* aRow1 = A + (long)(m0 + 16 + l16) * g.lda + hl * 16;

  const int nIter = g.K >> 5;

#if USE_TDM
  if (wave == 0) tdm_load_tile(Bt, lds_off(&smem[0][0]), g.ldb);
  for (int it = 0; it < nIter; ++it) {
    if (wave == 0) __builtin_amdgcn_s_wait_tensorcnt(0);
    __syncthreads();
    if (wave == 0 && (it + 1) < nIter)
      tdm_load_tile(Bt + (it + 1) * 32, lds_off(&smem[(it + 1) & 1][0]), g.ldb);
    const bf16_t* sb = &smem[it & 1][0];
    const int kb = it << 5;
#else
  for (int it = 0; it < nIter; ++it) {
    const int kb = it << 5;
    {  // cooperative copy: 128 rows x 64B, 32B per thread
      const int row = tid >> 1, seg = tid & 1;
      const bf16_t* src = Bt + (long)row * g.ldb + kb + seg * 16;
      bf16_t* dst = &smem[0][row * LDS_STRIDE + seg * 16];
      *(v8bf*)(dst)     = *(const v8bf*)(src);
      *(v8bf*)(dst + 8) = *(const v8bf*)(src + 8);
    }
    __syncthreads();
    const bf16_t* sb = &smem[0][0];
#endif

    __builtin_prefetch(aRow0 + kb + 128, 0, 1);
    __builtin_prefetch(aRow1 + kb + 128, 0, 1);

    Frag16 a0, a1;
    a0.h[0] = *(const v8bf*)(aRow0 + kb);
    a0.h[1] = *(const v8bf*)(aRow0 + kb + 8);
    a1.h[0] = *(const v8bf*)(aRow1 + kb);
    a1.h[1] = *(const v8bf*)(aRow1 + kb + 8);

    #pragma unroll
    for (int j = 0; j < 4; j++) {
      Frag16 bfr;
      const bf16_t* bp = sb + (n0loc + j * 16 + l16) * LDS_STRIDE + hl * 16;
      bfr.h[0] = *(const v8bf*)(bp);
      bfr.h[1] = *(const v8bf*)(bp + 8);
      acc[0][j] = wmma_bf16(a0, bfr, acc[0][j]);
      acc[1][j] = wmma_bf16(a1, bfr, acc[1][j]);
    }

#if !USE_TDM
    __syncthreads();
#endif
  }

  // epilogue
  #pragma unroll
  for (int i = 0; i < 2; i++) {
    #pragma unroll
    for (int j = 0; j < 4; j++) {
      const int nn = blockIdx.y * 128 + n0loc + j * 16 + l16;
      const float bv = g.bias ? g.bias[nn] : 0.0f;
      #pragma unroll
      for (int r = 0; r < 8; r++) {
        const int mm = m0 + i * 16 + hl * 8 + r;
        float v = acc[i][j][r] * g.scale + bv;
        if (g.relu) v = v > 0.0f ? v : 0.0f;
        long idx;
        if (g.mode == 0) {
          idx = (long)mm * g.ldc + nn;
        } else if (g.mode == 1) {
          const int bb = mm >> 9, ss = mm & 511;
          const int hh = nn >> 6, dd = nn & 63;
          idx = (((long)(bb * H_ + hh) * S_) + ss) * DH_ + dd;
        } else {
          const int bb = mm >> 9, ss = mm & 511;
          const int hh = nn >> 6, dd = nn & 63;
          idx = (((long)(bb * H_ + hh) * DH_) + dd) * S_ + ss;
        }
        if (g.Cf) g.Cf[z * g.sCf + idx] = v;
        if (g.Cb) g.Cb[z * g.sCb + idx] = (bf16_t)v;
      }
    }
  }

#if __has_builtin(__builtin_amdgcn_s_cluster_barrier)
  // NOP when not dispatched as a cluster (ClusterID==0, ISA §3.2)
  __builtin_amdgcn_s_cluster_barrier();
#endif
}

// ---------------------------------------------------------------------------
// fp32 [count][K][N]  ->  bf16 [count][N][K]   (weight transpose + convert)
// ---------------------------------------------------------------------------
__global__ __launch_bounds__(256)
void convT_kernel(const float* __restrict__ W, bf16_t* __restrict__ Wt,
                  int K, int N, long total) {
  long idx = (long)blockIdx.x * 256 + threadIdx.x;
  if (idx >= total) return;
  const long KN = (long)K * N;
  const long c   = idx / KN;
  const long rem = idx - c * KN;
  const int  k   = (int)(rem / N);
  const int  n   = (int)(rem - (long)k * N);
  Wt[c * KN + (long)n * K + k] = (bf16_t)W[idx];
}

// ---------------------------------------------------------------------------
// Embedding gather + sinusoidal positional encoding (writes f32 + bf16)
// ---------------------------------------------------------------------------
__global__ __launch_bounds__(256)
void embed_kernel(const int* __restrict__ tok, const float* __restrict__ emb,
                  float* __restrict__ x, bf16_t* __restrict__ xb) {
  const int row = blockIdx.x;
  const int s   = row & (S_ - 1);
  const int t   = threadIdx.x;
  const int id  = tok[row];
  #pragma unroll
  for (int e = 0; e < 2; e++) {
    const int d = t + e * 256;
    const float a = (float)(d & ~1);
    const float dv = __expf(a * (-9.210340371976184f / (float)D_));
    const float ang = (float)s * dv;
    const float pe  = (d & 1) ? __cosf(ang) : __sinf(ang);
    const float v = emb[(long)id * D_ + d] + pe;
    x[(long)row * D_ + d]  = v;
    xb[(long)row * D_ + d] = (bf16_t)v;
  }
}

// ---------------------------------------------------------------------------
// Masked softmax over keys: scores f32 [B*H*T, S] -> probs bf16
// ---------------------------------------------------------------------------
__global__ __launch_bounds__(256)
void softmax_kernel(const float* __restrict__ S, bf16_t* __restrict__ P,
                    const int* __restrict__ keyTok, int causal) {
  __shared__ float red[256];
  const int row = blockIdx.x;
  const int q   = row & (T_ - 1);
  const int b   = row >> 12;
  const float* sr = S + (long)row * S_;
  bf16_t*      pr = P + (long)row * S_;
  const int t = threadIdx.x;

  float v[2]; bool msk[2];
  #pragma unroll
  for (int e = 0; e < 2; e++) {
    const int k = t + e * 256;
    bool m = (keyTok[b * S_ + k] == 0);
    if (causal) m = m || (k > q);
    msk[e] = m;
    v[e] = m ? -3.0e38f : sr[k];
  }
  float mx = fmaxf(v[0], v[1]);
  red[t] = mx; __syncthreads();
  for (int o = 128; o > 0; o >>= 1) {
    if (t < o) red[t] = fmaxf(red[t], red[t + o]);
    __syncthreads();
  }
  mx = red[0]; __syncthreads();

  const float e0 = msk[0] ? 0.0f : __expf(v[0] - mx);
  const float e1 = msk[1] ? 0.0f : __expf(v[1] - mx);
  red[t] = e0 + e1; __syncthreads();
  for (int o = 128; o > 0; o >>= 1) {
    if (t < o) red[t] += red[t + o];
    __syncthreads();
  }
  const float sum = red[0];
  const float inv = sum > 0.0f ? 1.0f / sum : 0.0f;
  pr[t]       = (bf16_t)(e0 * inv);
  pr[t + 256] = (bf16_t)(e1 * inv);
}

// ---------------------------------------------------------------------------
// x = LayerNorm(x + a) * g + b   (writes f32 x in place + bf16 copy)
// ---------------------------------------------------------------------------
__global__ __launch_bounds__(256)
void add_ln_kernel(float* __restrict__ x, const float* __restrict__ a,
                   const float* __restrict__ gam, const float* __restrict__ bet,
                   bf16_t* __restrict__ xb) {
  __shared__ float red[256];
  const int row = blockIdx.x;
  const int t   = threadIdx.x;
  float* xr       = x + (long)row * D_;
  const float* ar = a + (long)row * D_;

  float v0 = xr[t]       + ar[t];
  float v1 = xr[t + 256] + ar[t + 256];
  red[t] = v0 + v1; __syncthreads();
  for (int o = 128; o > 0; o >>= 1) { if (t < o) red[t] += red[t + o]; __syncthreads(); }
  const float mean = red[0] * (1.0f / (float)D_); __syncthreads();

  const float d0 = v0 - mean, d1 = v1 - mean;
  red[t] = d0 * d0 + d1 * d1; __syncthreads();
  for (int o = 128; o > 0; o >>= 1) { if (t < o) red[t] += red[t + o]; __syncthreads(); }
  const float rstd = rsqrtf(red[0] * (1.0f / (float)D_) + 1e-5f);

  const float o0 = d0 * rstd * gam[t]       + bet[t];
  const float o1 = d1 * rstd * gam[t + 256] + bet[t + 256];
  xr[t] = o0; xr[t + 256] = o1;
  bf16_t* xbr = xb + (long)row * D_;
  xbr[t] = (bf16_t)o0; xbr[t + 256] = (bf16_t)o1;
}

// ---------------------------------------------------------------------------
// Host orchestration
// ---------------------------------------------------------------------------
extern "C" void kernel_launch(void* const* d_in, const int* in_sizes, int n_in,
                              void* d_out, int out_size, void* d_ws, size_t ws_size,
                              hipStream_t stream) {
  (void)in_sizes; (void)n_in; (void)out_size; (void)ws_size;

  const int*   src        = (const int*)  d_in[0];
  const int*   tgt        = (const int*)  d_in[1];
  const float* emb_src    = (const float*)d_in[2];
  const float* emb_tgt    = (const float*)d_in[3];
  const float* enc_attn_w = (const float*)d_in[4];
  const float* enc_attn_b = (const float*)d_in[5];
  const float* enc_ff_w1  = (const float*)d_in[6];
  const float* enc_ff_b1  = (const float*)d_in[7];
  const float* enc_ff_w2  = (const float*)d_in[8];
  const float* enc_ff_b2  = (const float*)d_in[9];
  const float* enc_ln_g   = (const float*)d_in[10];
  const float* enc_ln_b   = (const float*)d_in[11];
  const float* dec_attn_w = (const float*)d_in[12];
  const float* dec_attn_b = (const float*)d_in[13];
  const float* dec_ff_w1  = (const float*)d_in[14];
  const float* dec_ff_b1  = (const float*)d_in[15];
  const float* dec_ff_w2  = (const float*)d_in[16];
  const float* dec_ff_b2  = (const float*)d_in[17];
  const float* dec_ln_g   = (const float*)d_in[18];
  const float* dec_ln_bb  = (const float*)d_in[19];
  const float* fc_w       = (const float*)d_in[20];
  const float* fc_b       = (const float*)d_in[21];
  float* out = (float*)d_out;

  char* p = (char*)d_ws;
  auto alloc = [&](size_t bytes) -> void* {
    void* r = (void*)p;
    p += (bytes + 255) & ~(size_t)255;
    return r;
  };
  bf16_t* wT_enc_attn = (bf16_t*)alloc((size_t)L_ * 4 * D_ * D_ * 2);
  bf16_t* wT_dec_attn = (bf16_t*)alloc((size_t)L_ * 8 * D_ * D_ * 2);
  bf16_t* wT_eff1     = (bf16_t*)alloc((size_t)L_ * HID_ * D_ * 2);
  bf16_t* wT_eff2     = (bf16_t*)alloc((size_t)L_ * D_ * HID_ * 2);
  bf16_t* wT_dff1     = (bf16_t*)alloc((size_t)L_ * HID_ * D_ * 2);
  bf16_t* wT_dff2     = (bf16_t*)alloc((size_t)L_ * D_ * HID_ * 2);
  bf16_t* wT_fc       = (bf16_t*)alloc((size_t)VT_ * D_ * 2);
  float*  enc_x  = (float*) alloc((size_t)MROWS * D_ * 4);
  bf16_t* enc_xb = (bf16_t*)alloc((size_t)MROWS * D_ * 2);
  float*  dec_x  = (float*) alloc((size_t)MROWS * D_ * 4);
  bf16_t* dec_xb = (bf16_t*)alloc((size_t)MROWS * D_ * 2);
  bf16_t* qb     = (bf16_t*)alloc((size_t)MROWS * D_ * 2);   // [B,H,S,DH]
  bf16_t* kb2    = (bf16_t*)alloc((size_t)MROWS * D_ * 2);   // [B,H,S,DH]
  bf16_t* vbt    = (bf16_t*)alloc((size_t)MROWS * D_ * 2);   // [B,H,DH,S]
  float*  scores = (float*) alloc((size_t)B_ * H_ * T_ * S_ * 4);
  bf16_t* probs  = (bf16_t*)alloc((size_t)B_ * H_ * T_ * S_ * 2);
  bf16_t* ctxb   = (bf16_t*)alloc((size_t)MROWS * D_ * 2);
  float*  tmp    = (float*) alloc((size_t)MROWS * D_ * 4);
  bf16_t* hidb   = (bf16_t*)alloc((size_t)MROWS * HID_ * 2);

  auto convT = [&](const float* W, bf16_t* Wt, int K, int N, long count) {
    const long total = count * (long)K * N;
    const int blocks = (int)((total + 255) / 256);
    convT_kernel<<<blocks, 256, 0, stream>>>(W, Wt, K, N, total);
  };
  convT(enc_attn_w, wT_enc_attn, D_, D_, (long)L_ * 4);
  convT(dec_attn_w, wT_dec_attn, D_, D_, (long)L_ * 8);
  convT(enc_ff_w1,  wT_eff1, D_, HID_, L_);
  convT(enc_ff_w2,  wT_eff2, HID_, D_, L_);
  convT(dec_ff_w1,  wT_dff1, D_, HID_, L_);
  convT(dec_ff_w2,  wT_dff2, HID_, D_, L_);
  convT(fc_w,       wT_fc,   D_, VT_, 1);

  embed_kernel<<<MROWS, 256, 0, stream>>>(src, emb_src, enc_x, enc_xb);
  embed_kernel<<<MROWS, 256, 0, stream>>>(tgt, emb_tgt, dec_x, dec_xb);

  auto gemm = [&](const bf16_t* A, int lda, long sA,
                  const bf16_t* Bt, int ldb, long sB,
                  const float* bias, float* Cf, long sCf, bf16_t* Cb, long sCb,
                  int M, int N, int K, int ldc, int mode, int relu, float scale,
                  int batch) {
    GemmArgs ga;
    ga.A = A; ga.Bt = Bt; ga.bias = bias; ga.Cf = Cf; ga.Cb = Cb;
    ga.sA = sA; ga.sB = sB; ga.sCf = sCf; ga.sCb = sCb;
    ga.K = K; ga.lda = lda; ga.ldb = ldb; ga.ldc = ldc;
    ga.mode = mode; ga.relu = relu; ga.scale = scale;
    if (mode != 3 && (N % 128) == 0) {
      dim3 grid(M / 128, N / 128, batch);
      gemm_bf16_wmma_lds<<<grid, dim3(256), 0, stream>>>(ga);
    } else {
      dim3 grid(M / 128, N / 64, batch);
      gemm_bf16_wmma<<<grid, dim3(256), 0, stream>>>(ga);
    }
  };

  auto attention = [&](const bf16_t* xq_b, const bf16_t* xkv_b,
                       const bf16_t* Wt4, const float* b4,
                       const int* keyTok, int causal, float* outTmp) {
    const bf16_t* WqT = Wt4;
    const bf16_t* WkT = Wt4 + (long)D_ * D_;
    const bf16_t* WvT = Wt4 + 2L * D_ * D_;
    const bf16_t* WoT = Wt4 + 3L * D_ * D_;
    const float *bq = b4, *bk = b4 + D_, *bv = b4 + 2 * D_, *bo = b4 + 3 * D_;

    gemm(xq_b,  D_, 0, WqT, D_, 0, bq, nullptr, 0, qb,  0, MROWS, D_, D_, 0, 1, 0, 1.0f, 1);
    gemm(xkv_b, D_, 0, WkT, D_, 0, bk, nullptr, 0, kb2, 0, MROWS, D_, D_, 0, 1, 0, 1.0f, 1);
    gemm(xkv_b, D_, 0, WvT, D_, 0, bv, nullptr, 0, vbt, 0, MROWS, D_, D_, 0, 2, 0, 1.0f, 1);
    gemm(qb, DH_, (long)S_ * DH_, kb2, DH_, (long)S_ * DH_, nullptr,
         scores, (long)T_ * S_, nullptr, 0,
         T_, S_, DH_, S_, 0, 0, 0.125f, B_ * H_);
    softmax_kernel<<<B_ * H_ * T_, 256, 0, stream>>>(scores, probs, keyTok, causal);
    gemm(probs, S_, (long)T_ * S_, vbt, S_, (long)DH_ * S_, nullptr,
         nullptr, 0, ctxb, 0,
         T_, DH_, S_, 0, 3, 0, 1.0f, B_ * H_);
    gemm(ctxb, D_, 0, WoT, D_, 0, bo, outTmp, 0, nullptr, 0,
         MROWS, D_, D_, D_, 0, 0, 1.0f, 1);
  };

  auto ffn = [&](bf16_t* x_b, const bf16_t* W1T, const float* b1,
                 const bf16_t* W2T, const float* b2, float* outTmp) {
    gemm(x_b, D_, 0, W1T, D_, 0, b1, nullptr, 0, hidb, 0,
         MROWS, HID_, D_, HID_, 0, 1, 1.0f, 1);
    gemm(hidb, HID_, 0, W2T, HID_, 0, b2, outTmp, 0, nullptr, 0,
         MROWS, D_, HID_, D_, 0, 0, 1.0f, 1);
  };

  // ---------------- encoder ----------------
  for (int l = 0; l < L_; l++) {
    attention(enc_xb, enc_xb,
              wT_enc_attn + (long)l * 4 * D_ * D_,
              enc_attn_b + (long)l * 4 * D_,
              src, 0, tmp);
    add_ln_kernel<<<MROWS, 256, 0, stream>>>(enc_x, tmp,
        enc_ln_g + (long)(l * 2 + 0) * D_, enc_ln_b + (long)(l * 2 + 0) * D_, enc_xb);
    ffn(enc_xb, wT_eff1 + (long)l * HID_ * D_, enc_ff_b1 + (long)l * HID_,
        wT_eff2 + (long)l * D_ * HID_, enc_ff_b2 + (long)l * D_, tmp);
    add_ln_kernel<<<MROWS, 256, 0, stream>>>(enc_x, tmp,
        enc_ln_g + (long)(l * 2 + 1) * D_, enc_ln_b + (long)(l * 2 + 1) * D_, enc_xb);
  }

  // ---------------- decoder ----------------
  for (int l = 0; l < L_; l++) {
    attention(dec_xb, dec_xb,
              wT_dec_attn + (long)(l * 2 + 0) * 4 * D_ * D_,
              dec_attn_b + (long)(l * 2 + 0) * 4 * D_,
              tgt, 1, tmp);
    add_ln_kernel<<<MROWS, 256, 0, stream>>>(dec_x, tmp,
        dec_ln_g + (long)(l * 3 + 0) * D_, dec_ln_bb + (long)(l * 3 + 0) * D_, dec_xb);
    attention(dec_xb, enc_xb,
              wT_dec_attn + (long)(l * 2 + 1) * 4 * D_ * D_,
              dec_attn_b + (long)(l * 2 + 1) * 4 * D_,
              src, 0, tmp);
    add_ln_kernel<<<MROWS, 256, 0, stream>>>(dec_x, tmp,
        dec_ln_g + (long)(l * 3 + 1) * D_, dec_ln_bb + (long)(l * 3 + 1) * D_, dec_xb);
    ffn(dec_xb, wT_dff1 + (long)l * HID_ * D_, dec_ff_b1 + (long)l * HID_,
        wT_dff2 + (long)l * D_ * HID_, dec_ff_b2 + (long)l * D_, tmp);
    add_ln_kernel<<<MROWS, 256, 0, stream>>>(dec_x, tmp,
        dec_ln_g + (long)(l * 3 + 2) * D_, dec_ln_bb + (long)(l * 3 + 2) * D_, dec_xb);
  }

  // ---------------- logits ----------------
  gemm(dec_xb, D_, 0, wT_fc, D_, 0, fc_b, out, 0, nullptr, 0,
       MROWS, VT_, D_, VT_, 0, 0, 1.0f, 1);
}
